// RNNMemoryCell_25434796327774
// MI455X (gfx1250) — compile-verified, matrix operand order checked by reading, not run on hardware
//
#include <hip/hip_runtime.h>
#include <hip/hip_bf16.h>
#include <math.h>

// Problem constants (match reference)
#define Bb   2
#define Nn   1024
#define DIMm 1024
#define Hh   8
#define DHh  64
#define DIi  512
#define Kc   4
#define EPSf 1.1920929e-07f

typedef __bf16 bf16_t;
typedef __bf16 bf16x16 __attribute__((ext_vector_type(16)));
typedef __bf16 bf16x8  __attribute__((ext_vector_type(8)));
typedef float  floatx8 __attribute__((ext_vector_type(8)));

// gfx1250 async global->LDS path (ASYNCcnt), with graceful fallback.
#if defined(__gfx1250__) && __has_builtin(__builtin_amdgcn_global_load_async_to_lds_b32)
#define ASYNC_LDS_OK 1
#else
#define ASYNC_LDS_OK 0
#endif

#if ASYNC_LDS_OK
typedef __attribute__((address_space(1))) int gint_as1;
typedef __attribute__((address_space(3))) int lint_as3;
#endif

__device__ __forceinline__ void async_ld_b32(const float* g, float* l) {
#if ASYNC_LDS_OK
  __builtin_amdgcn_global_load_async_to_lds_b32(
      (gint_as1*)g, (lint_as3*)l, /*offset=*/0, /*cpol=*/0);
#else
  *l = *g;
#endif
}

__device__ __forceinline__ void wait_async_lds() {
#if ASYNC_LDS_OK
#if __has_builtin(__builtin_amdgcn_s_wait_asynccnt)
  __builtin_amdgcn_s_wait_asynccnt(0);
#else
  asm volatile("s_wait_asynccnt 0x0" ::: "memory");
#endif
#endif
}

// ---------------------------------------------------------------------------
// 1. RMSNorm over DIM. One block per (b,t) row.
// ---------------------------------------------------------------------------
__global__ void rmsnorm_kernel(const float* __restrict__ x,
                               const float* __restrict__ w,
                               float* __restrict__ xa) {
  __shared__ float red[256];
  const int row = blockIdx.x;                       // 0..B*N-1
  const float* xr = x + (size_t)row * DIMm;
  float ss = 0.f;
  for (int d = threadIdx.x; d < DIMm; d += 256) { float v = xr[d]; ss += v * v; }
  red[threadIdx.x] = ss;
  __syncthreads();
  for (int s = 128; s > 0; s >>= 1) {
    if (threadIdx.x < s) red[threadIdx.x] += red[threadIdx.x + s];
    __syncthreads();
  }
  const float scale = rsqrtf(red[0] * (1.0f / DIMm) + EPSf);
  float* out = xa + (size_t)row * DIMm;
  for (int d = threadIdx.x; d < DIMm; d += 256) out[d] = xr[d] * scale * w[d];
}

// ---------------------------------------------------------------------------
// 2. Fused depthwise conv (K=4, taps t-2..t+1) for q,k,v paths -> bf16.
// ---------------------------------------------------------------------------
__global__ void dwconv_kernel(const float* __restrict__ xa,
                              const float* __restrict__ cq,
                              const float* __restrict__ ck,
                              const float* __restrict__ cv,
                              bf16_t* __restrict__ qi,
                              bf16_t* __restrict__ ki,
                              bf16_t* __restrict__ vi) {
  const int row = blockIdx.x;                       // b*N + t
  const int b = row / Nn, t = row % Nn;
  const int d0 = threadIdx.x * 4;
  for (int dd = 0; dd < 4; ++dd) {
    const int d = d0 + dd;
    float aq = 0.f, ak = 0.f, av = 0.f;
    for (int j = 0; j < Kc; ++j) {
      const int tt = t - 2 + j;
      if (tt < 0 || tt >= Nn) continue;
      const float xv = xa[((size_t)(b * Nn + tt)) * DIMm + d];
      aq += xv * cq[d * Kc + j];
      ak += xv * ck[d * Kc + j];
      av += xv * cv[d * Kc + j];
    }
    const size_t o = (size_t)row * DIMm + d;
    qi[o] = (bf16_t)aq; ki[o] = (bf16_t)ak; vi[o] = (bf16_t)av;
  }
}

// ---------------------------------------------------------------------------
// 3. fp32 -> bf16 transposed weight conversion. in: rows x cols -> out: cols x rows
// ---------------------------------------------------------------------------
__global__ void transpose_to_bf16(const float* __restrict__ in,
                                  bf16_t* __restrict__ out,
                                  int rows, int cols) {
  int idx = blockIdx.x * blockDim.x + threadIdx.x;
  const int total = rows * cols;
  for (; idx < total; idx += gridDim.x * blockDim.x) {
    const int r = idx / cols, c = idx % cols;
    out[(size_t)c * rows + r] = (bf16_t)in[idx];
  }
}

// ---------------------------------------------------------------------------
// 4. bf16 WMMA GEMM, 32x32 macro-tile per wave (2x2 register blocking):
//    C(MxN, f32) = A(MxK, bf16 row-major) * Bt(NxK, bf16)^T
//    Per K-step of 32: 2 A fragments + 2 B fragments -> 4 WMMAs.
// ---------------------------------------------------------------------------
__device__ __forceinline__ bf16x16 load_frag(const bf16_t* row) {
  bf16x8 lo = *(const bf16x8*)(row);
  bf16x8 hi = *(const bf16x8*)(row + 16);
  bf16x16 f;
  for (int i = 0; i < 8; ++i) { f[i] = lo[i]; f[i + 8] = hi[i]; }
  return f;
}

__global__ void gemm_bf16_wmma(const bf16_t* __restrict__ A,
                               const bf16_t* __restrict__ Bt,
                               float* __restrict__ C,
                               int M, int N, int Kd) {
  const int tileN = blockIdx.x * 32;
  const int tileM = blockIdx.y * 32;
  const int lane = threadIdx.x & 31;
  const int half = lane >> 4;                       // 0 or 1
  const int r    = lane & 15;
  const bf16_t* arow0 = A  + (size_t)(tileM + r) * Kd + half * 8;
  const bf16_t* arow1 = arow0 + (size_t)16 * Kd;
  const bf16_t* brow0 = Bt + (size_t)(tileN + r) * Kd + half * 8;
  const bf16_t* brow1 = brow0 + (size_t)16 * Kd;

  floatx8 acc00 = {}, acc01 = {}, acc10 = {}, acc11 = {};
  for (int k0 = 0; k0 < Kd; k0 += 32) {
    __builtin_prefetch(arow0 + k0 + 64, 0, 3);
    __builtin_prefetch(brow0 + k0 + 64, 0, 3);
    bf16x16 a0 = load_frag(arow0 + k0);
    bf16x16 a1 = load_frag(arow1 + k0);
    bf16x16 b0 = load_frag(brow0 + k0);
    bf16x16 b1 = load_frag(brow1 + k0);
    acc00 = __builtin_amdgcn_wmma_f32_16x16x32_bf16(false, a0, false, b0, (short)0, acc00, false, false);
    acc01 = __builtin_amdgcn_wmma_f32_16x16x32_bf16(false, a0, false, b1, (short)0, acc01, false, false);
    acc10 = __builtin_amdgcn_wmma_f32_16x16x32_bf16(false, a1, false, b0, (short)0, acc10, false, false);
    acc11 = __builtin_amdgcn_wmma_f32_16x16x32_bf16(false, a1, false, b1, (short)0, acc11, false, false);
  }
  // C layout per tile: VGPR i -> (M = i + 8*half, N = lane&15)
  const int c0 = tileN + r, c1 = tileN + 16 + r;
  for (int i = 0; i < 8; ++i) {
    const int r0 = tileM + i + half * 8;
    const int r1 = r0 + 16;
    C[(size_t)r0 * N + c0] = acc00[i];
    C[(size_t)r0 * N + c1] = acc01[i];
    C[(size_t)r1 * N + c0] = acc10[i];
    C[(size_t)r1 * N + c1] = acc11[i];
  }
}

// ---------------------------------------------------------------------------
// 5. Per-head RMSNorm (64-wide rows), in place.
// ---------------------------------------------------------------------------
__global__ void mh_rmsnorm_kernel(float* __restrict__ q,
                                  const float* __restrict__ gamma) {
  __shared__ float red[64];
  const int rh = blockIdx.x;                        // row*H + h
  const int row = rh / Hh, h = rh % Hh;
  const int d = threadIdx.x;
  const size_t o = (size_t)row * DIi + h * DHh + d;
  const float v = q[o];
  red[d] = v * v;
  __syncthreads();
  for (int s = 32; s > 0; s >>= 1) {
    if (d < s) red[d] += red[d + s];
    __syncthreads();
  }
  const float rms   = sqrtf(red[0]) * 0.125f;       // * DH^-0.5
  const float scale = 1.f / fmaxf(rms, 1e-8f);
  q[o] = v * scale * gamma[h * DHh + d];
}

// ---------------------------------------------------------------------------
// 6. Gate vectors: sigmoid(xa @ W + b) for lr / decay / mom, stored (b*H, N).
//    One block per (b,t); 24 dot products of length 1024, tree-reduced.
// ---------------------------------------------------------------------------
__global__ void gates_kernel(const float* __restrict__ xa,
                             const float* __restrict__ lw, const float* __restrict__ lb,
                             const float* __restrict__ dw, const float* __restrict__ db,
                             const float* __restrict__ mw, const float* __restrict__ mb,
                             float* __restrict__ lr, float* __restrict__ de,
                             float* __restrict__ mo) {
  __shared__ float part[256 * 24];
  const int row = blockIdx.x;                       // b*N + t
  const int b = row / Nn, t = row % Nn;
  const float* xr = xa + (size_t)row * DIMm;
  float p[24];
  for (int g = 0; g < 24; ++g) p[g] = 0.f;
  for (int d = threadIdx.x; d < DIMm; d += 256) {
    const float xv = xr[d];
    for (int h = 0; h < Hh; ++h) {
      p[h]      += xv * lw[d * Hh + h];
      p[8 + h]  += xv * dw[d * Hh + h];
      p[16 + h] += xv * mw[d * Hh + h];
    }
  }
  for (int g = 0; g < 24; ++g) part[threadIdx.x * 24 + g] = p[g];
  __syncthreads();
  for (int s = 128; s > 0; s >>= 1) {
    if (threadIdx.x < s)
      for (int g = 0; g < 24; ++g)
        part[threadIdx.x * 24 + g] += part[(threadIdx.x + s) * 24 + g];
    __syncthreads();
  }
  if (threadIdx.x < 24) {
    const int g = threadIdx.x;
    const float s = part[g];
    const int h = g & 7, which = g >> 3;
    const float bias = (which == 0) ? lb[h] : (which == 1) ? db[h] : mb[h];
    const float val = 1.f / (1.f + __expf(-(s + bias)));
    float* dst = (which == 0) ? lr : (which == 1) ? de : mo;
    dst[(size_t)(b * Hh + h) * Nn + t] = val;
  }
}

// ---------------------------------------------------------------------------
// 7. Sequential gated scan.  One workgroup per (b,h); 256 threads each own a
//    4x4 tile of the 64x64 Z / S states in registers.
//      y_t = S_{t-1} q_t ;  Z_t = mom_t Z_{t-1} - v_t k_t^T ;
//      S_t = decay_t S_{t-1} - lr_t Z_t
//    q/k/v double-buffered in LDS via async global->LDS loads (ASYNCcnt);
//    y-row reduction via ds-permute shuffles; one barrier per step.
// ---------------------------------------------------------------------------
__global__ void scan_kernel(const float* __restrict__ Q,
                            const float* __restrict__ Km,
                            const float* __restrict__ V,
                            const float* __restrict__ LR,
                            const float* __restrict__ DE,
                            const float* __restrict__ MO,
                            bf16_t* __restrict__ Y) {
  __shared__ float qkv[2][3][64];                   // double buffer
  __shared__ float glr[Nn], gde[Nn], gmo[Nn];       // per-step gates, 12 KB
  const int bh = blockIdx.x;
  const int b = bh / Hh, h = bh % Hh;
  const int tid = threadIdx.x;
  const int tr = tid >> 4, tc = tid & 15;           // 4-row / 4-col tile coords

  // Preload all gates for this sequence into LDS (hit LDS, not HBM, per step).
  for (int i = tid; i < Nn; i += 256) {
    glr[i] = LR[(size_t)bh * Nn + i];
    gde[i] = DE[(size_t)bh * Nn + i];
    gmo[i] = MO[(size_t)bh * Nn + i];
  }

  float Z[4][4], S[4][4];
  for (int i = 0; i < 4; ++i)
    for (int j = 0; j < 4; ++j) { Z[i][j] = 0.f; S[i][j] = 0.f; }

  // Prologue: issue async load of step 0 into buffer 0.
  {
    const size_t base0 = (size_t)(b * Nn) * DIi + h * DHh;
    if (tid < 64) {
      async_ld_b32(Q  + base0 + tid, &qkv[0][0][tid]);
      async_ld_b32(Km + base0 + tid, &qkv[0][1][tid]);
      async_ld_b32(V  + base0 + tid, &qkv[0][2][tid]);
    }
  }

  for (int t = 0; t < Nn; ++t) {
    const int s = t & 1;
    wait_async_lds();                               // data for step t landed
    __syncthreads();                                // visible to all waves;
                                                    // prior step's reads done
    // Issue step t+1 into the other buffer; overlaps with compute below.
    if (t + 1 < Nn && tid < 64) {
      const size_t basen = (size_t)(b * Nn + t + 1) * DIi + h * DHh;
      async_ld_b32(Q  + basen + tid, &qkv[s ^ 1][0][tid]);
      async_ld_b32(Km + basen + tid, &qkv[s ^ 1][1][tid]);
      async_ld_b32(V  + basen + tid, &qkv[s ^ 1][2][tid]);
    }

    const float* qs = qkv[s][0];
    const float* ks = qkv[s][1];
    const float* vs = qkv[s][2];
    const float lr  = glr[t];
    const float dec = gde[t];
    const float mo  = gmo[t];
    const size_t base = (size_t)(b * Nn + t) * DIi + h * DHh;

    // y = S_{t-1} q_t : per-thread 4x4 dot, then reduce across the 16 lanes
    // (tc = 0..15) of each half-wave with shuffles (no barrier needed).
    float yv[4];
    for (int i = 0; i < 4; ++i) {
      float acc = 0.f;
      for (int j = 0; j < 4; ++j) acc += S[i][j] * qs[4 * tc + j];
      acc += __shfl_xor(acc, 1, 16);
      acc += __shfl_xor(acc, 2, 16);
      acc += __shfl_xor(acc, 4, 16);
      acc += __shfl_xor(acc, 8, 16);
      yv[i] = acc;
    }
    if (tc == 0) {
      for (int i = 0; i < 4; ++i) Y[base + 4 * tr + i] = (bf16_t)yv[i];
    }

    // State update (uses ks/vs of buffer s; next iteration's top barrier
    // orders the overwrite of this buffer two steps from now).
    for (int i = 0; i < 4; ++i) {
      const float nv = -vs[4 * tr + i];
      for (int j = 0; j < 4; ++j) {
        const float g = nv * ks[4 * tc + j];
        Z[i][j] = mo * Z[i][j] + g;
        S[i][j] = dec * S[i][j] - lr * Z[i][j];
      }
    }
  }
}

// ---------------------------------------------------------------------------
// Launch
// ---------------------------------------------------------------------------
extern "C" void kernel_launch(void* const* d_in, const int* in_sizes, int n_in,
                              void* d_out, int out_size, void* d_ws, size_t ws_size,
                              hipStream_t stream) {
  const float* x      = (const float*)d_in[0];
  const float* w_rms  = (const float*)d_in[1];
  const float* wq     = (const float*)d_in[2];
  const float* wk     = (const float*)d_in[3];
  const float* wv     = (const float*)d_in[4];
  const float* wo     = (const float*)d_in[5];
  const float* conv_q = (const float*)d_in[6];
  const float* conv_k = (const float*)d_in[7];
  const float* conv_v = (const float*)d_in[8];
  const float* gamma_q = (const float*)d_in[9];
  const float* gamma_k = (const float*)d_in[10];
  const float* lr_w   = (const float*)d_in[11];
  const float* lr_b   = (const float*)d_in[12];
  const float* de_w   = (const float*)d_in[13];
  const float* de_b   = (const float*)d_in[14];
  const float* mo_w   = (const float*)d_in[15];
  const float* mo_b   = (const float*)d_in[16];
  float* out = (float*)d_out;

  // Workspace carving (256B aligned)
  char* ws = (char*)d_ws;
  size_t off = 0;
  auto carve = [&](size_t bytes) -> char* {
    char* p = ws + off;
    off += (bytes + 255) & ~(size_t)255;
    return p;
  };
  const size_t ROWS = (size_t)Bb * Nn;              // 2048
  float*  xa   = (float*) carve(ROWS * DIMm * 4);
  bf16_t* qi   = (bf16_t*)carve(ROWS * DIMm * 2);
  bf16_t* ki   = (bf16_t*)carve(ROWS * DIMm * 2);
  bf16_t* vi   = (bf16_t*)carve(ROWS * DIMm * 2);
  bf16_t* wqt  = (bf16_t*)carve((size_t)DIi * DIMm * 2);
  bf16_t* wkt  = (bf16_t*)carve((size_t)DIi * DIMm * 2);
  bf16_t* wvt  = (bf16_t*)carve((size_t)DIi * DIMm * 2);
  bf16_t* wot  = (bf16_t*)carve((size_t)DIMm * DIi * 2);
  float*  qbuf = (float*) carve(ROWS * DIi * 4);
  float*  kbuf = (float*) carve(ROWS * DIi * 4);
  float*  vbuf = (float*) carve(ROWS * DIi * 4);
  float*  lrb  = (float*) carve((size_t)Bb * Hh * Nn * 4);
  float*  deb  = (float*) carve((size_t)Bb * Hh * Nn * 4);
  float*  mob  = (float*) carve((size_t)Bb * Hh * Nn * 4);
  bf16_t* ybf  = (bf16_t*)carve(ROWS * DIi * 2);
  (void)ws_size; (void)n_in; (void)in_sizes; (void)out_size;

  // 1. RMSNorm
  rmsnorm_kernel<<<dim3(ROWS), dim3(256), 0, stream>>>(x, w_rms, xa);

  // 2. Depthwise convs -> bf16 activations
  dwconv_kernel<<<dim3(ROWS), dim3(256), 0, stream>>>(xa, conv_q, conv_k, conv_v, qi, ki, vi);

  // 3. Weight conversions (transposed bf16)
  transpose_to_bf16<<<dim3(512), dim3(256), 0, stream>>>(wq, wqt, DIMm, DIi);
  transpose_to_bf16<<<dim3(512), dim3(256), 0, stream>>>(wk, wkt, DIMm, DIi);
  transpose_to_bf16<<<dim3(512), dim3(256), 0, stream>>>(wv, wvt, DIMm, DIi);
  transpose_to_bf16<<<dim3(512), dim3(256), 0, stream>>>(wo, wot, DIi, DIMm);

  // 4. Projections (WMMA, 32x32 macro-tiles): (2048x1024) x (1024x512)
  {
    dim3 grid(DIi / 32, ROWS / 32);
    gemm_bf16_wmma<<<grid, dim3(32), 0, stream>>>(qi, wqt, qbuf, (int)ROWS, DIi, DIMm);
    gemm_bf16_wmma<<<grid, dim3(32), 0, stream>>>(ki, wkt, kbuf, (int)ROWS, DIi, DIMm);
    gemm_bf16_wmma<<<grid, dim3(32), 0, stream>>>(vi, wvt, vbuf, (int)ROWS, DIi, DIMm);
  }

  // 5. Per-head RMSNorm on q and k
  mh_rmsnorm_kernel<<<dim3(ROWS * Hh), dim3(64), 0, stream>>>(qbuf, gamma_q);
  mh_rmsnorm_kernel<<<dim3(ROWS * Hh), dim3(64), 0, stream>>>(kbuf, gamma_k);

  // 6. Gates
  gates_kernel<<<dim3(ROWS), dim3(256), 0, stream>>>(xa, lr_w, lr_b, de_w, de_b,
                                                     mo_w, mo_b, lrb, deb, mob);

  // 7. Recurrent scan: 16 sequences, state in registers, async LDS pipeline
  scan_kernel<<<dim3(Bb * Hh), dim3(256), 0, stream>>>(qbuf, kbuf, vbuf,
                                                       lrb, deb, mob, ybf);

  // 8. Output projection (WMMA): (2048x512) x (512x1024) -> d_out fp32
  {
    dim3 grid(DIMm / 32, ROWS / 32);
    gemm_bf16_wmma<<<grid, dim3(32), 0, stream>>>(ybf, wot, out, (int)ROWS, DIMm, DIi);
  }
}
